// RNN_11768210391210
// MI455X (gfx1250) — matrix operand order repaired and checked by reading, not compiled, not run on hardware
//
#include <hip/hip_runtime.h>

// 3-layer LSTM (B=256, T=512, H=128, IN=64), PyTorch gate order i,f,g,o.
// One workgroup per 16 batch rows runs the whole time loop:
//  - fused [Whh | Wih] resident in LDS as f16 (264 KB < 320 KB CDNA5 WGP LDS)
//  - gates = bias + [h | x_t] @ [Whh | Wih]^T via v_wmma_f32_16x16x32_f16
//  - c in registers; h ping-pongs through double-buffered LDS tiles
//  - ONE s_barrier per timestep (parity double-buffering removes WAR hazards)
//  - layers 1/2 run in-place on the final `out` region (read (b,t) before write)

typedef _Float16 v4h  __attribute__((ext_vector_type(4)));
typedef _Float16 v8h  __attribute__((ext_vector_type(8)));
typedef _Float16 v16h __attribute__((ext_vector_type(16)));
typedef float    v8f  __attribute__((ext_vector_type(8)));

#define H_      128
#define T_      512
#define B_      256
#define NG4H    512   // 4*H rows of the fused weight matrix
#define STRIDE  264   // f16 elems per LDS row: 528 B = 132 dwords -> conflict-minimal banks

extern __shared__ __align__(16) _Float16 smem[];

__device__ __forceinline__ float fast_sigm(float x) {
    return __builtin_amdgcn_rcpf(1.0f + __expf(-x));
}
__device__ __forceinline__ float fast_tanh(float x) {
    // correct limits: x->+inf => 1-2*rcp(inf)=1 ; x->-inf => 1-2*rcp(1)=-1
    return 1.0f - 2.0f * __builtin_amdgcn_rcpf(__expf(2.0f * x) + 1.0f);
}

#define VCAT16(lo, hi) __builtin_shufflevector(lo, hi, 0,1,2,3,4,5,6,7,8,9,10,11,12,13,14,15)

template <int IN_SHIFT>
__global__ __launch_bounds__(256, 1) void lstm_layer_kernel(
    const float* inPtr,
    const float* __restrict__ Whh, const float* __restrict__ Wih,
    const float* __restrict__ bih, const float* __restrict__ bhh,
    float* outBuf, float* __restrict__ hN, float* __restrict__ cN,
    float* __restrict__ hTop, int isLast)
{
    constexpr int IN_DIM = 1 << IN_SHIFT;
    constexpr int NKC    = (H_ + IN_DIM) >> 5;   // K chunks of 32 (6 or 8)

    _Float16* w16  = smem;                        // [512][STRIDE] fused [Whh|Wih] f16
    _Float16* actA = smem + NG4H * STRIDE;        // [16][STRIDE] buf0: h cols [0,128), x cols [128,..)
    _Float16* actB = actA + 16 * STRIDE;          // [16][STRIDE] buf1

    const int tid  = threadIdx.x;
    const int wave = tid >> 5;            // 0..7, owns h-columns [16w,16w+16)
    const int lane = tid & 31;
    const int ln   = lane & 15;
    const int hi   = lane >> 4;           // WMMA half-wave selector
    const int b0   = blockIdx.x << 4;     // 16 batch rows per workgroup
    const int jcol = (wave << 4) + ln;    // this lane's output column j in [0,128)

    // ---- stage weights (f32 global -> f16 LDS), coalesced ----
    for (int idx = tid; idx < NG4H * H_; idx += 256) {
        int n = idx >> 7, k = idx & (H_ - 1);
        w16[n * STRIDE + k] = (_Float16)Whh[idx];
    }
    for (int idx = tid; idx < (NG4H << IN_SHIFT); idx += 256) {
        int n = idx >> IN_SHIFT, k = idx & (IN_DIM - 1);
        w16[n * STRIDE + H_ + k] = (_Float16)Wih[idx];
    }
    // ---- h0 = 0 in both parity buffers ----
    for (int idx = tid; idx < 16 * H_; idx += 256) {
        int r = idx >> 7, k = idx & (H_ - 1);
        actA[r * STRIDE + k] = (_Float16)0.f;
        actB[r * STRIDE + k] = (_Float16)0.f;
    }

    // per-lane gate biases (broadcast across the 8 accumulator rows)
    const float bI = bih[jcol]        + bhh[jcol];
    const float bF = bih[H_   + jcol] + bhh[H_   + jcol];
    const float bG = bih[2*H_ + jcol] + bhh[2*H_ + jcol];
    const float bO = bih[3*H_ + jcol] + bhh[3*H_ + jcol];

    v8f cReg, hReg;
    #pragma unroll
    for (int r = 0; r < 8; ++r) { cReg[r] = 0.f; hReg[r] = 0.f; }

    __syncthreads();

    const int aOff = hi ? 16 : 0;    // A frag: K runs {0-15} / {16-31}
    const int bOff = hi ? 16 : 0;    // B frag: K runs {0-15} / {16-31}

    for (int t = 0; t < T_; ++t) {
        _Float16* rd = (t & 1) ? actB : actA;   // holds h_{t-1}; receives x_t
        _Float16* wr = (t & 1) ? actA : actB;   // receives h_t for step t+1

        // ---- stage x_t: float4 global loads -> ds_store_b64 f16 ----
        constexpr int XC4 = (16 << IN_SHIFT) >> 2;
        #pragma unroll
        for (int i = 0; i < XC4 / 256; ++i) {
            int idx = tid + i * 256;
            int r  = idx >> (IN_SHIFT - 2);
            int k4 = idx & ((IN_DIM >> 2) - 1);
            float4 v = *(const float4*)&inPtr[(((size_t)(b0 + r)) * T_ + t) * IN_DIM + (k4 << 2)];
            v4h p;
            p[0] = (_Float16)v.x; p[1] = (_Float16)v.y;
            p[2] = (_Float16)v.z; p[3] = (_Float16)v.w;
            *(v4h*)&rd[r * STRIDE + H_ + (k4 << 2)] = p;
        }
        __syncthreads();  // the single per-step barrier

        v8f accI, accF, accG, accO;
        #pragma unroll
        for (int r = 0; r < 8; ++r) { accI[r]=bI; accF[r]=bF; accG[r]=bG; accO[r]=bO; }

        #pragma unroll
        for (int kc = 0; kc < NKC; ++kc) {
            const int kb = kc << 5;
            // issue ALL 10 ds_load_b128 of this chunk before any WMMA,
            // so the scheduler uses partial dscnt waits / overlaps with XDL
            const _Float16* ap  = rd + ln * STRIDE + kb + aOff;
            const _Float16* bp0 = w16 + jcol * STRIDE + kb + bOff;      // gate i
            const _Float16* bp1 = bp0 + 1 * H_ * STRIDE;                // gate f
            const _Float16* bp2 = bp0 + 2 * H_ * STRIDE;                // gate g
            const _Float16* bp3 = bp0 + 3 * H_ * STRIDE;                // gate o
            v8h a0  = *(const v8h*)ap;       v8h a1  = *(const v8h*)(ap  + 8);
            v8h q0l = *(const v8h*)bp0;      v8h q0h = *(const v8h*)(bp0 + 8);
            v8h q1l = *(const v8h*)bp1;      v8h q1h = *(const v8h*)(bp1 + 8);
            v8h q2l = *(const v8h*)bp2;      v8h q2h = *(const v8h*)(bp2 + 8);
            v8h q3l = *(const v8h*)bp3;      v8h q3h = *(const v8h*)(bp3 + 8);
            v16h Af = VCAT16(a0, a1);
            v16h B0 = VCAT16(q0l, q0h);
            v16h B1 = VCAT16(q1l, q1h);
            v16h B2 = VCAT16(q2l, q2h);
            v16h B3 = VCAT16(q3l, q3h);
            accI = __builtin_amdgcn_wmma_f32_16x16x32_f16(false, Af, false, B0, (short)0, accI, false, false);
            accF = __builtin_amdgcn_wmma_f32_16x16x32_f16(false, Af, false, B1, (short)0, accF, false, false);
            accG = __builtin_amdgcn_wmma_f32_16x16x32_f16(false, Af, false, B2, (short)0, accG, false, false);
            accO = __builtin_amdgcn_wmma_f32_16x16x32_f16(false, Af, false, B3, (short)0, accO, false, false);
        }

        // ---- activations + state update (C/D layout: VGPR r -> row hi*8+r, col jcol) ----
        #pragma unroll
        for (int r = 0; r < 8; ++r) {
            float iv = fast_sigm(accI[r]);
            float fv = fast_sigm(accF[r]);
            float gv = fast_tanh(accG[r]);
            float ov = fast_sigm(accO[r]);
            float cv = fv * cReg[r] + iv * gv;
            cReg[r] = cv;
            float hv = ov * fast_tanh(cv);
            hReg[r] = hv;
            int row = (hi << 3) + r;
            wr[row * STRIDE + jcol] = (_Float16)hv;                      // h_t for step t+1
            outBuf[(((size_t)(b0 + row)) * T_ + t) * H_ + jcol] = hv;    // layer output
        }
        // writes go to the opposite parity buffer; the next step's barrier
        // publishes them before they are read — no trailing barrier needed
    }

    #pragma unroll
    for (int r = 0; r < 8; ++r) {
        int row = (hi << 3) + r;
        size_t o = (size_t)(b0 + row) * H_ + jcol;
        hN[o] = hReg[r];
        cN[o] = cReg[r];
        if (isLast) hTop[o] = hReg[r];
    }
}

extern "C" void kernel_launch(void* const* d_in, const int* in_sizes, int n_in,
                              void* d_out, int out_size, void* d_ws, size_t ws_size,
                              hipStream_t stream)
{
    (void)in_sizes; (void)n_in; (void)out_size; (void)d_ws; (void)ws_size;
    const float* x = (const float*)d_in[0];
    float* out = (float*)d_out;

    const size_t OUT_N = (size_t)B_ * T_ * H_;       // 16,777,216
    float* outR = out;                               // out [B,T,H]
    float* hTop = out + OUT_N;                       // h_top [B,H]
    float* hN   = hTop + (size_t)B_ * H_;            // h_n [3,B,H]
    float* cN   = hN + 3ull * B_ * H_;               // c_n [3,B,H]

    const int smemBytes = (NG4H + 32) * STRIDE * 2;  // 287,232 B < 320 KB
    hipFuncSetAttribute((const void*)lstm_layer_kernel<6>,
                        hipFuncAttributeMaxDynamicSharedMemorySize, smemBytes);
    hipFuncSetAttribute((const void*)lstm_layer_kernel<7>,
                        hipFuncAttributeMaxDynamicSharedMemorySize, smemBytes);

    for (int layer = 0; layer < 3; ++layer) {
        const float* Wih = (const float*)d_in[1 + 4*layer + 0];
        const float* Whh = (const float*)d_in[1 + 4*layer + 1];
        const float* bih = (const float*)d_in[1 + 4*layer + 2];
        const float* bhh = (const float*)d_in[1 + 4*layer + 3];
        float* hNl = hN + (size_t)layer * B_ * H_;
        float* cNl = cN + (size_t)layer * B_ * H_;
        if (layer == 0) {
            lstm_layer_kernel<6><<<B_ / 16, 256, smemBytes, stream>>>(
                x, Whh, Wih, bih, bhh, outR, hNl, cNl, hTop, 0);
        } else {
            lstm_layer_kernel<7><<<B_ / 16, 256, smemBytes, stream>>>(
                outR, Whh, Wih, bih, bhh, outR, hNl, cNl, hTop,
                (layer == 2) ? 1 : 0);
        }
    }
}
